// MultiHeadAttention_51307679318159
// MI455X (gfx1250) — compile-verified, matrix operand order checked by reading, not run on hardware
//
#include <hip/hip_runtime.h>

// MI455X / gfx1250 multi-head attention forward.
// GEMMs: v_wmma_f32_16x16x32_bf16 (fp32 accumulate). Softmax: fp32.
// bf16 tile staging uses GLOBAL_LOAD_ASYNC_TO_LDS_B64 (ASYNCcnt path);
// fp32->bf16 staging uses float4 loads + v_cvt_pk_bf16_f32 + ds_store_b64.
// All staging loops are compile-time unrolled (4 iterations, no exec-mask loops).

typedef __attribute__((ext_vector_type(16))) __bf16 bf16x16;
typedef __attribute__((ext_vector_type(4)))  __bf16 bf16x4;
typedef __attribute__((ext_vector_type(8)))  float  f32x8;
typedef __attribute__((ext_vector_type(4)))  float  f32x4;

#define B_   2
#define S_   2048
#define D_   1024
#define H_   16
#define DK_  64
#define BH_  (B_ * H_)

#define BM   64
#define BN   64
#define BK   32
#define NT   128          // 4 waves per workgroup (wave32)
#define LDSS (BK + 8)     // LDS row stride (bf16 elems); 80B rows keep 8B align
#define CHUNKS   (BM * (BK / 4))      // 512 4-elem chunks per 64x32 tile
#define STAGE_IT (CHUNKS / NT)        // 4 iterations per thread

__device__ __forceinline__ f32x8 wmma_bf16(bf16x16 a, bf16x16 b, f32x8 c) {
  // (neg_a, A, neg_b, B, c_mod, C, reuse_a, reuse_b)
  return __builtin_amdgcn_wmma_f32_16x16x32_bf16(false, a, false, b, (short)0, c,
                                                 false, false);
}

// 16x32 bf16 A/B fragment from LDS tile (row-major, stride LDSS).
// lanes 0-15: rows 0-15, K = 0..7 (VGPR0-3) and 16..23 (VGPR4-7);
// lanes 16-31: same rows, K = 8..15 and 24..31.
__device__ __forceinline__ bf16x16 frag_ld(const __bf16* base, int row, int lane) {
  const __bf16* p = base + (row + (lane & 15)) * LDSS + ((lane >> 4) << 3);
  bf16x16 f;
#pragma unroll
  for (int i = 0; i < 8; ++i) {
    f[i]     = p[i];
    f[i + 8] = p[i + 16];
  }
  return f;
}

// Async copy 8 bytes (4 bf16) global -> LDS, tracked with ASYNCcnt.
// LDS aperture maps generic addr[31:0] to the LDS byte offset (ISA 10.2).
__device__ __forceinline__ void async_cp_b64(__bf16* lds_dst, const __bf16* gsrc) {
  unsigned lds_off = (unsigned)(unsigned long long)(uintptr_t)lds_dst;
  unsigned long long gaddr = (unsigned long long)(uintptr_t)gsrc;
  asm volatile("global_load_async_to_lds_b64 %0, %1, off"
               :: "v"(lds_off), "v"(gaddr)
               : "memory");
}

__device__ __forceinline__ void wait_async0() {
  asm volatile("s_wait_asynccnt 0x0" ::: "memory");
}

// Stage a 64xBK fp32 tile into LDS as bf16 (straight-line, 4 iterations).
__device__ __forceinline__ void stage_f32(__bf16* lds, const float* src,
                                          size_t ld, int tid) {
#pragma unroll
  for (int it = 0; it < STAGE_IT; ++it) {
    int idx = tid + it * NT;
    int r = idx >> 3;           // BK/4 == 8 chunks per row
    int c = (idx & 7) << 2;
    f32x4 v = *(const f32x4*)(src + (size_t)r * ld + c);
    *(bf16x4*)(lds + r * LDSS + c) = __builtin_convertvector(v, bf16x4);
  }
}

// Stage a 64xBK bf16 tile into LDS via async global->LDS copies (4 issues).
__device__ __forceinline__ void stage_bf16_async(__bf16* lds, const __bf16* src,
                                                 size_t ld, int tid) {
#pragma unroll
  for (int it = 0; it < STAGE_IT; ++it) {
    int idx = tid + it * NT;
    int r = idx >> 3;
    int c = (idx & 7) << 2;
    async_cp_b64(lds + r * LDSS + c, src + (size_t)r * ld + c);
  }
}

// ---------------------------------------------------------------------------
// Kernel 1: QKV projection, specialized per matrix (MAT=0 Q, 1 K, 2 V).
// x[B*S, D] @ W^T + b.  Q,K -> bf16 [B,H,S,DK]; V -> transposed bf16 [B,H,DK,S].
// ---------------------------------------------------------------------------
template <int MAT>
__global__ __launch_bounds__(NT)
void qkv_proj_kernel(const float* __restrict__ x, const float* __restrict__ W,
                     const float* __restrict__ bias, __bf16* __restrict__ dst) {
  __shared__ __bf16 lA[BM * LDSS];
  __shared__ __bf16 lB[BN * LDSS];

  const int tid  = threadIdx.x;
  const int lane = tid & 31;
  const int wid  = tid >> 5;
  const int m0   = blockIdx.x * BM;   // row index into [0, B*S)
  const int n0   = blockIdx.y * BN;   // output feature

  const int wm = (wid >> 1) * 32;
  const int wn = (wid & 1) * 32;

  f32x8 acc[2][2] = {};

  for (int k0 = 0; k0 < D_; k0 += BK) {
    stage_f32(lA, x + (size_t)m0 * D_ + k0, D_, tid);
    stage_f32(lB, W + (size_t)n0 * D_ + k0, D_, tid);
    if (k0 + BK < D_) {  // prefetch next weight K-tile
      __builtin_prefetch(W + (size_t)(n0 + (tid >> 1)) * D_ + k0 + BK, 0, 3);
    }
    __syncthreads();
    bf16x16 a0 = frag_ld(lA, wm,      lane);
    bf16x16 a1 = frag_ld(lA, wm + 16, lane);
    bf16x16 b0 = frag_ld(lB, wn,      lane);
    bf16x16 b1 = frag_ld(lB, wn + 16, lane);
    acc[0][0] = wmma_bf16(a0, b0, acc[0][0]);
    acc[0][1] = wmma_bf16(a0, b1, acc[0][1]);
    acc[1][0] = wmma_bf16(a1, b0, acc[1][0]);
    acc[1][1] = wmma_bf16(a1, b1, acc[1][1]);
    __syncthreads();
  }

  const int cn = lane & 15;
  const int r8 = (lane >> 4) * 8;
#pragma unroll
  for (int ti = 0; ti < 2; ++ti)
#pragma unroll
    for (int tj = 0; tj < 2; ++tj)
#pragma unroll
      for (int r = 0; r < 8; ++r) {
        int gm = m0 + wm + ti * 16 + r8 + r;        // b*S + s
        int gn = n0 + wn + tj * 16 + cn;            // feature -> h*DK + dk
        float v = acc[ti][tj][r] + bias[gn];
        int b = gm >> 11, s = gm & (S_ - 1);
        int h = gn >> 6,  dk = gn & (DK_ - 1);
        if (MAT == 2)  // V stored transposed [B,H,DK,S]
          dst[((size_t)(b * H_ + h) * DK_ + dk) * S_ + s] = (__bf16)v;
        else           // Q/K stored [B,H,S,DK]
          dst[((size_t)(b * H_ + h) * S_ + s) * DK_ + dk] = (__bf16)v;
      }
}

// ---------------------------------------------------------------------------
// Kernel 2: scores = Q @ K^T / sqrt(DK), per (b,h).  fp32 into attn region.
// Q/K tiles staged with async global->LDS copies.
// ---------------------------------------------------------------------------
__global__ __launch_bounds__(NT)
void scores_kernel(const __bf16* __restrict__ qf, const __bf16* __restrict__ kf,
                   float* __restrict__ attn) {
  __shared__ __bf16 lA[BM * LDSS];
  __shared__ __bf16 lB[BN * LDSS];

  const int tid  = threadIdx.x;
  const int lane = tid & 31;
  const int wid  = tid >> 5;
  const int bh   = blockIdx.z;
  const int m0   = blockIdx.x * BM;   // q rows
  const int n0   = blockIdx.y * BN;   // k rows

  const __bf16* Q  = qf + (size_t)bh * S_ * DK_;
  const __bf16* Km = kf + (size_t)bh * S_ * DK_;
  float* out = attn + (size_t)bh * S_ * S_;

  const int wm = (wid >> 1) * 32;
  const int wn = (wid & 1) * 32;

  f32x8 acc[2][2] = {};

#pragma unroll
  for (int k0 = 0; k0 < DK_; k0 += BK) {
    stage_bf16_async(lA, Q  + (size_t)m0 * DK_ + k0, DK_, tid);
    stage_bf16_async(lB, Km + (size_t)n0 * DK_ + k0, DK_, tid);
    wait_async0();
    __syncthreads();
    bf16x16 a0 = frag_ld(lA, wm,      lane);
    bf16x16 a1 = frag_ld(lA, wm + 16, lane);
    bf16x16 b0 = frag_ld(lB, wn,      lane);
    bf16x16 b1 = frag_ld(lB, wn + 16, lane);
    acc[0][0] = wmma_bf16(a0, b0, acc[0][0]);
    acc[0][1] = wmma_bf16(a0, b1, acc[0][1]);
    acc[1][0] = wmma_bf16(a1, b0, acc[1][0]);
    acc[1][1] = wmma_bf16(a1, b1, acc[1][1]);
    __syncthreads();
  }

  const int cn = lane & 15;
  const int r8 = (lane >> 4) * 8;
  const float scale = 0.125f;  // 1/sqrt(64)
#pragma unroll
  for (int ti = 0; ti < 2; ++ti)
#pragma unroll
    for (int tj = 0; tj < 2; ++tj)
#pragma unroll
      for (int r = 0; r < 8; ++r) {
        int q = m0 + wm + ti * 16 + r8 + r;
        int k = n0 + wn + tj * 16 + cn;
        out[(size_t)q * S_ + k] = acc[ti][tj][r] * scale;
      }
}

// ---------------------------------------------------------------------------
// Kernel 3: row softmax over attn, fp32 in place, float4-vectorized.
// One 256-thread block per row of 2048 (= 512 float4, 2 per thread).
// ---------------------------------------------------------------------------
__global__ __launch_bounds__(256)
void softmax_kernel(float* __restrict__ attn) {
  __shared__ float red[256];
  const int t = threadIdx.x;
  f32x4* p = (f32x4*)(attn + (size_t)blockIdx.x * S_);

  float m = -3.402823466e+38f;
#pragma unroll
  for (int it = 0; it < S_ / 4 / 256; ++it) {
    f32x4 v = p[t + it * 256];
    m = fmaxf(m, fmaxf(fmaxf(v.x, v.y), fmaxf(v.z, v.w)));
  }
  red[t] = m;
  __syncthreads();
  for (int s = 128; s > 0; s >>= 1) {
    if (t < s) red[t] = fmaxf(red[t], red[t + s]);
    __syncthreads();
  }
  m = red[0];
  __syncthreads();

  float sum = 0.0f;
#pragma unroll
  for (int it = 0; it < S_ / 4 / 256; ++it) {
    f32x4 v = p[t + it * 256];
    v.x = __expf(v.x - m);
    v.y = __expf(v.y - m);
    v.z = __expf(v.z - m);
    v.w = __expf(v.w - m);
    p[t + it * 256] = v;
    sum += (v.x + v.y) + (v.z + v.w);
  }
  red[t] = sum;
  __syncthreads();
  for (int s = 128; s > 0; s >>= 1) {
    if (t < s) red[t] += red[t + s];
    __syncthreads();
  }
  float inv = 1.0f / red[0];
#pragma unroll
  for (int it = 0; it < S_ / 4 / 256; ++it) {
    f32x4 v = p[t + it * 256];
    v.x *= inv; v.y *= inv; v.z *= inv; v.w *= inv;
    p[t + it * 256] = v;
  }
}

// ---------------------------------------------------------------------------
// Kernel 4: head output = attn @ V, per (b,h).  attn fp32 -> bf16 staged in LDS,
// V held transposed [DK,S] (async-staged).  Output merged bf16 [B,S,D].
// ---------------------------------------------------------------------------
__global__ __launch_bounds__(NT)
void av_kernel(const float* __restrict__ attn, const __bf16* __restrict__ vT,
               __bf16* __restrict__ yf) {
  __shared__ __bf16 lA[BM * LDSS];
  __shared__ __bf16 lB[BN * LDSS];

  const int tid  = threadIdx.x;
  const int lane = tid & 31;
  const int wid  = tid >> 5;
  const int bh   = blockIdx.z;
  const int b    = bh / H_;
  const int h    = bh % H_;
  const int m0   = blockIdx.x * BM;   // q rows; BN == DK covers all columns

  const float*  A  = attn + (size_t)bh * S_ * S_;
  const __bf16* Bv = vT + (size_t)bh * DK_ * S_;

  const int wm = (wid >> 1) * 32;
  const int wn = (wid & 1) * 32;

  f32x8 acc[2][2] = {};

  for (int k0 = 0; k0 < S_; k0 += BK) {
    stage_bf16_async(lB, Bv + k0, S_, tid);            // bf16 copy (async)
    stage_f32(lA, A + (size_t)m0 * S_ + k0, S_, tid);  // fp32 -> bf16
    wait_async0();
    __syncthreads();
    bf16x16 a0 = frag_ld(lA, wm,      lane);
    bf16x16 a1 = frag_ld(lA, wm + 16, lane);
    bf16x16 b0 = frag_ld(lB, wn,      lane);
    bf16x16 b1 = frag_ld(lB, wn + 16, lane);
    acc[0][0] = wmma_bf16(a0, b0, acc[0][0]);
    acc[0][1] = wmma_bf16(a0, b1, acc[0][1]);
    acc[1][0] = wmma_bf16(a1, b0, acc[1][0]);
    acc[1][1] = wmma_bf16(a1, b1, acc[1][1]);
    __syncthreads();
  }

  const int cn = lane & 15;
  const int r8 = (lane >> 4) * 8;
#pragma unroll
  for (int ti = 0; ti < 2; ++ti)
#pragma unroll
    for (int tj = 0; tj < 2; ++tj)
#pragma unroll
      for (int r = 0; r < 8; ++r) {
        int s  = m0 + wm + ti * 16 + r8 + r;
        int dk = wn + tj * 16 + cn;
        yf[(size_t)(b * S_ + s) * D_ + h * DK_ + dk] = (__bf16)acc[ti][tj][r];
      }
}

// ---------------------------------------------------------------------------
// Kernel 5: out = y @ Wo^T + bo  (fp32 output).  Y tile async-staged.
// ---------------------------------------------------------------------------
__global__ __launch_bounds__(NT)
void oproj_kernel(const __bf16* __restrict__ yf, const float* __restrict__ Wo,
                  const float* __restrict__ bo, float* __restrict__ out) {
  __shared__ __bf16 lA[BM * LDSS];
  __shared__ __bf16 lB[BN * LDSS];

  const int tid  = threadIdx.x;
  const int lane = tid & 31;
  const int wid  = tid >> 5;
  const int m0   = blockIdx.x * BM;
  const int n0   = blockIdx.y * BN;

  const int wm = (wid >> 1) * 32;
  const int wn = (wid & 1) * 32;

  f32x8 acc[2][2] = {};

  for (int k0 = 0; k0 < D_; k0 += BK) {
    stage_bf16_async(lA, yf + (size_t)m0 * D_ + k0, D_, tid);
    stage_f32(lB, Wo + (size_t)n0 * D_ + k0, D_, tid);
    if (k0 + BK < D_) {
      __builtin_prefetch(Wo + (size_t)(n0 + (tid >> 1)) * D_ + k0 + BK, 0, 3);
    }
    wait_async0();
    __syncthreads();
    bf16x16 a0 = frag_ld(lA, wm,      lane);
    bf16x16 a1 = frag_ld(lA, wm + 16, lane);
    bf16x16 b0 = frag_ld(lB, wn,      lane);
    bf16x16 b1 = frag_ld(lB, wn + 16, lane);
    acc[0][0] = wmma_bf16(a0, b0, acc[0][0]);
    acc[0][1] = wmma_bf16(a0, b1, acc[0][1]);
    acc[1][0] = wmma_bf16(a1, b0, acc[1][0]);
    acc[1][1] = wmma_bf16(a1, b1, acc[1][1]);
    __syncthreads();
  }

  const int cn = lane & 15;
  const int r8 = (lane >> 4) * 8;
#pragma unroll
  for (int ti = 0; ti < 2; ++ti)
#pragma unroll
    for (int tj = 0; tj < 2; ++tj)
#pragma unroll
      for (int r = 0; r < 8; ++r) {
        int gm = m0 + wm + ti * 16 + r8 + r;
        int gn = n0 + wn + tj * 16 + cn;
        out[(size_t)gm * D_ + gn] = acc[ti][tj][r] + bo[gn];
      }
}

// ---------------------------------------------------------------------------
extern "C" void kernel_launch(void* const* d_in, const int* in_sizes, int n_in,
                              void* d_out, int out_size, void* d_ws, size_t ws_size,
                              hipStream_t stream) {
  const float* x  = (const float*)d_in[0];
  const float* Wq = (const float*)d_in[1];
  const float* bq = (const float*)d_in[2];
  const float* Wk = (const float*)d_in[3];
  const float* bk = (const float*)d_in[4];
  const float* Wv = (const float*)d_in[5];
  const float* bv = (const float*)d_in[6];
  const float* Wo = (const float*)d_in[7];
  const float* bo = (const float*)d_in[8];

  float* out  = (float*)d_out;
  float* attn = out + (size_t)B_ * S_ * D_;   // second output region

  const size_t E = (size_t)B_ * S_ * D_;      // 4M elems per bf16 buffer
  __bf16* qf = (__bf16*)d_ws;
  __bf16* kf = qf + E;
  __bf16* vT = kf + E;
  __bf16* yf = vT + E;

  dim3 blk(NT);
  dim3 gproj((B_ * S_) / BM, D_ / BN);
  qkv_proj_kernel<0><<<gproj, blk, 0, stream>>>(x, Wq, bq, qf);
  qkv_proj_kernel<1><<<gproj, blk, 0, stream>>>(x, Wk, bk, kf);
  qkv_proj_kernel<2><<<gproj, blk, 0, stream>>>(x, Wv, bv, vT);
  scores_kernel<<<dim3(S_ / BM, S_ / BN, BH_), blk, 0, stream>>>(qf, kf, attn);
  softmax_kernel<<<dim3(BH_ * S_), dim3(256), 0, stream>>>(attn);
  av_kernel<<<dim3(S_ / BM, 1, BH_), blk, 0, stream>>>(attn, vT, yf);
  oproj_kernel<<<dim3((B_ * S_) / BM, D_ / BN), blk, 0, stream>>>(yf, Wo, bo, out);
}
